// Cluster_35089882808435
// MI455X (gfx1250) — matrix operand order repaired and checked by reading, not compiled, not run on hardware
//
#include <hip/hip_runtime.h>

typedef __attribute__((ext_vector_type(16))) __bf16 v16bf;
typedef __attribute__((ext_vector_type(8)))  float  v8f;
typedef __attribute__((ext_vector_type(4)))  float  f4v;

#define N_NODES 8192
#define N_COMM  16
#define D_DIM   64
#define EPSF    1e-7f

// ---------- Kernel A: manifold distances + softmax -> piT (bf16, layout [community][node]) ----------
__global__ __launch_bounds__(256) void pi_kernel(const float* __restrict__ emb,
                                                 const float* __restrict__ comm,
                                                 __bf16* __restrict__ piT,
                                                 float* __restrict__ wsAcc) {
  __shared__ float cS[N_COMM * D_DIM];
  int tid = threadIdx.x;
  for (int t = tid; t < N_COMM * D_DIM; t += 256) cS[t] = comm[t];
  if (blockIdx.x == 0 && tid < 128) wsAcc[tid] = 0.0f;   // zero 64 accumulator pairs each launch
  __syncthreads();

  int i = blockIdx.x * 256 + tid;   // grid = 32 blocks * 256 = 8192 exactly
  float x[D_DIM];
  float d2[N_COMM];
#pragma unroll
  for (int k = 0; k < N_COMM; ++k) d2[k] = 0.0f;

  // Euclidean (curvature 0): dist^2 = ||x - c||^2
  {
    const float* xp = emb + (size_t)i * D_DIM;
#pragma unroll
    for (int d = 0; d < D_DIM; ++d) x[d] = xp[d];
    for (int k = 0; k < N_COMM; ++k) {
      float s = 0.0f;
#pragma unroll
      for (int d = 0; d < D_DIM; ++d) { float df = x[d] - cS[k*D_DIM+d]; s = fmaf(df, df, s); }
      d2[k] += s;
    }
  }
  // Lorentz (curvature -1 -> kk = 1): dist = acosh(max(-(x.c - 2*x0*c0), 1+eps))
  {
    const float* xp = emb + (size_t)(N_NODES + i) * D_DIM;
#pragma unroll
    for (int d = 0; d < D_DIM; ++d) x[d] = xp[d];
    for (int k = 0; k < N_COMM; ++k) {
      float ip = 0.0f;
#pragma unroll
      for (int d = 0; d < D_DIM; ++d) ip = fmaf(x[d], cS[k*D_DIM+d], ip);
      float lip = ip - 2.0f * x[0] * cS[k*D_DIM];
      float arg = fmaxf(-lip, 1.0f + EPSF);
      float dd = acoshf(arg);
      d2[k] = fmaf(dd, dd, d2[k]);
    }
  }
  // Sphere (curvature +1): dist = acos(clip(x.c))
  {
    const float* xp = emb + (size_t)(2*N_NODES + i) * D_DIM;
#pragma unroll
    for (int d = 0; d < D_DIM; ++d) x[d] = xp[d];
    for (int k = 0; k < N_COMM; ++k) {
      float ip = 0.0f;
#pragma unroll
      for (int d = 0; d < D_DIM; ++d) ip = fmaf(x[d], cS[k*D_DIM+d], ip);
      float cv = fminf(fmaxf(ip, -1.0f + EPSF), 1.0f - EPSF);
      float dd = acosf(cv);
      d2[k] = fmaf(dd, dd, d2[k]);
    }
  }

  float dn[N_COMM];
#pragma unroll
  for (int k = 0; k < N_COMM; ++k) dn[k] = sqrtf(d2[k]);
  float m = dn[0];
#pragma unroll
  for (int k = 1; k < N_COMM; ++k) m = fmaxf(m, dn[k]);
  float e[N_COMM]; float ssum = 0.0f;
#pragma unroll
  for (int k = 0; k < N_COMM; ++k) { e[k] = expf(dn[k] - m); ssum += e[k]; }
  float inv = 1.0f / ssum;
#pragma unroll
  for (int k = 0; k < N_COMM; ++k)
    piT[(size_t)k * N_NODES + i] = (__bf16)(e[k] * inv);
}

// ---------- Kernel B: tiled Y = R @ pi via v_wmma_f32_16x16x32_bf16, fused reductions ----------
// grid = (512 row-tiles, 4), block = 256 (8 waves). Wave handles a segment of 256 columns,
// software-pipelined (double-buffered) so next stage's loads fly under current cvt+WMMA.
__global__ __launch_bounds__(256) void gemm_kernel(const float* __restrict__ R,
                                                   const __bf16* __restrict__ piT,
                                                   float* __restrict__ wsAcc) {
  int tid  = threadIdx.x;
  int wave = tid >> 5;
  int lane = tid & 31;
  int hi   = lane >> 4;        // 0: lanes 0-15, 1: lanes 16-31
  int l16  = lane & 15;
  int i0   = blockIdx.x * 16;
  int seg  = blockIdx.y * 8 + wave;   // 0..31

  const float*  rowp = R   + (size_t)(i0 + l16) * N_NODES;  // A row: M = l16
  const __bf16* bp   = piT + (size_t)l16 * N_NODES;         // B col: N = l16
  int cbase = hi * 16;   // A K-offset, contiguous (K 0-15 vs 16-31), matches B
  int kb    = hi * 16;   // B K-offset, contiguous (K 0-15 vs 16-31), matches A

  int jstart = seg * 256;

  // stage-0 prologue loads
  f4v f0 = *(const f4v*)(rowp + jstart + cbase);
  f4v f1 = *(const f4v*)(rowp + jstart + cbase + 4);
  f4v f2 = *(const f4v*)(rowp + jstart + cbase + 8);
  f4v f3 = *(const f4v*)(rowp + jstart + cbase + 12);
  v16bf bmat = *(const v16bf*)(bp + jstart + kb);

  v8f acc = {};
#pragma unroll
  for (int jj = 0; jj < 256; jj += 32) {
    // issue next stage's loads BEFORE consuming the current stage
    f4v g0 = f0, g1 = f1, g2 = f2, g3 = f3;
    v16bf bn = bmat;
    if (jj + 32 < 256) {
      int c1 = jstart + jj + 32 + cbase;
      g0 = *(const f4v*)(rowp + c1);
      g1 = *(const f4v*)(rowp + c1 + 4);
      g2 = *(const f4v*)(rowp + c1 + 8);
      g3 = *(const f4v*)(rowp + c1 + 12);
      bn = *(const v16bf*)(bp + jstart + jj + 32 + kb);
    }
    v16bf a;
#pragma unroll
    for (int e = 0; e < 4; ++e) {
      a[e]      = (__bf16)f0[e];
      a[4 + e]  = (__bf16)f1[e];
      a[8 + e]  = (__bf16)f2[e];
      a[12 + e] = (__bf16)f3[e];
    }
    acc = __builtin_amdgcn_wmma_f32_16x16x32_bf16(false, a, false, bmat,
                                                  (short)0, acc, false, false);
    f0 = g0; f1 = g1; f2 = g2; f3 = g3; bmat = bn;
  }

  // D layout: lane<16 -> (M=r, N=lane); lane>=16 -> (M=8+r, N=lane-16)
  // intra_part = sum pi[i,k]*Y[i,k]; inter_part = sum Y[i,k]  (s == 1: softmax rows sum to 1)
  const __bf16* pcol = piT + (size_t)l16 * N_NODES + i0 + hi * 8;
  float intra = 0.0f, inter = 0.0f;
#pragma unroll
  for (int r = 0; r < 8; ++r) {
    float y = acc[r];
    intra = fmaf((float)pcol[r], y, intra);
    inter += y;
  }
#pragma unroll
  for (int off = 16; off >= 1; off >>= 1) {
    intra += __shfl_xor(intra, off, 32);
    inter += __shfl_xor(inter, off, 32);
  }
  __shared__ float red[16];
  if (lane == 0) { red[wave * 2] = intra; red[wave * 2 + 1] = inter; }
  __syncthreads();
  if (tid == 0) {
    float ia = 0.0f, ib = 0.0f;
#pragma unroll
    for (int w = 0; w < 8; ++w) { ia += red[2*w]; ib += red[2*w + 1]; }
    int slot = (blockIdx.x & 63) * 2;   // stripe atomics over 64 pairs
    atomicAdd(&wsAcc[slot], ia);
    atomicAdd(&wsAcc[slot + 1], ib);
  }
}

// ---------- Kernel C: finalize scalar ----------
__global__ void final_kernel(const float* __restrict__ wsAcc,
                             const float* __restrict__ alpha,
                             float* __restrict__ out) {
  if (threadIdx.x == 0) {
    float intra = 0.0f, inter = 0.0f;
    for (int t = 0; t < 64; ++t) { intra += wsAcc[2*t]; inter += wsAcc[2*t + 1]; }
    // intra / (K*N) = /131072 ; inter / (K*K*N) = /2097152
    out[0] = alpha[0] * intra * (1.0f / 131072.0f) - inter * (1.0f / 2097152.0f);
  }
}

extern "C" void kernel_launch(void* const* d_in, const int* in_sizes, int n_in,
                              void* d_out, int out_size, void* d_ws, size_t ws_size,
                              hipStream_t stream) {
  (void)in_sizes; (void)n_in; (void)out_size; (void)ws_size;
  const float* emb   = (const float*)d_in[0];   // (3, 8192, 64) f32
  const float* comm  = (const float*)d_in[1];   // (16, 64) f32
  const float* ricci = (const float*)d_in[2];   // (8192, 8192) f32
  const float* alpha = (const float*)d_in[3];   // scalar f32
  float* out   = (float*)d_out;
  float* wsAcc = (float*)d_ws;                  // 128 floats: 64 (intra,inter) pairs
  __bf16* piT  = (__bf16*)(wsAcc + 128);        // 16*8192 bf16 = 256KB, 32B-aligned

  pi_kernel<<<dim3(32), dim3(256), 0, stream>>>(emb, comm, piT, wsAcc);
  gemm_kernel<<<dim3(512, 4), dim3(256), 0, stream>>>(ricci, piT, wsAcc);
  final_kernel<<<dim3(1), dim3(32), 0, stream>>>(wsAcc, alpha, out);
}
